// GemmRSMultiNode_31997506355741
// MI455X (gfx1250) — matrix operand order repaired and checked by reading, not compiled, not run on hardware
//
#include <hip/hip_runtime.h>

typedef __bf16  bf16_t;
typedef bf16_t  bf16x16 __attribute__((ext_vector_type(16)));
typedef bf16_t  bf16x8  __attribute__((ext_vector_type(8)));
typedef bf16_t  bf16x4  __attribute__((ext_vector_type(4)));
typedef float   f32x8   __attribute__((ext_vector_type(8)));

#define M_DIM   4096
#define N_DIM   4096
#define KLOC    1024
#define RANKS   8
#define BM      128
#define BN      128
#define BK      32
#define BKP     40                      // padded LDS row stride (80B): 16B-aligned, bank-conflict-free b128
#define NSTEPS  (RANKS * (KLOC / BK))   // 256 k-steps of 32
#define TILES_M (M_DIM / BM)            // 32
#define TILES_N (N_DIM / BN)            // 32
#define STRIPE  16                      // N-tiles per L2 stripe

union FragU { bf16x16 v; bf16x8 h[2]; };

__global__ __launch_bounds__(256, 2)
void gemm_rs_bf16_wmma(const float* __restrict__ A,
                       const float* __restrict__ B,
                       float* __restrict__ C)
{
    __shared__ __align__(16) bf16_t As[2][BM * BKP];
    __shared__ __align__(16) bf16_t Bs[2][BN * BKP];

    const int tid  = threadIdx.x;
    const int lane = tid & 31;
    const int wave = tid >> 5;
    const int wm   = wave & 1;      // 2 waves along M (64 rows each)
    const int wn   = wave >> 1;     // 4 waves along N (32 cols each)

    // L2-friendly block swizzle: sweep all M-tiles within a 16-wide N stripe.
    const int bid    = (int)blockIdx.x;
    const int stripe = bid / (TILES_M * STRIPE);
    const int inr    = bid % (TILES_M * STRIPE);
    const int bm0    = (inr / STRIPE) * BM;
    const int bn0    = (stripe * STRIPE + (inr % STRIPE)) * BN;

    // global->LDS staging: each thread owns 4 float4 per tile
    const int lrow = tid >> 3;         // 0..31
    const int lcol = (tid & 7) << 2;   // 0,4,...,28

    // fragment lane decomposition (ISA 16-bit A / B layouts, wave32)
    const int fm = lane & 15;
    const int kg = lane >> 4;          // 0/1

    f32x8 acc[4][2];
#pragma unroll
    for (int i = 0; i < 4; ++i)
#pragma unroll
        for (int j = 0; j < 2; ++j)
            acc[i][j] = (f32x8){0.f,0.f,0.f,0.f,0.f,0.f,0.f,0.f};

    float4 ra[4], rb[4];

    auto gload = [&](int s) {
        const size_t rbase = (size_t)(s >> 5) * (size_t)M_DIM * KLOC;  // rank chunk
        const int    kk    = (s & 31) << 5;                            // k within rank
        const float* Ab = A + rbase + (size_t)bm0 * KLOC + kk;
        const float* Bb = B + rbase + (size_t)bn0 * KLOC + kk;
#pragma unroll
        for (int p = 0; p < 4; ++p) {
            ra[p] = *(const float4*)(Ab + (size_t)(lrow + p * 32) * KLOC + lcol);
            rb[p] = *(const float4*)(Bb + (size_t)(lrow + p * 32) * KLOC + lcol);
        }
    };

    auto lstore = [&](int buf) {
#pragma unroll
        for (int p = 0; p < 4; ++p) {
            const int row = lrow + p * 32;
            bf16x4 ha, hb;
            ha[0] = (bf16_t)ra[p].x; ha[1] = (bf16_t)ra[p].y;
            ha[2] = (bf16_t)ra[p].z; ha[3] = (bf16_t)ra[p].w;
            hb[0] = (bf16_t)rb[p].x; hb[1] = (bf16_t)rb[p].y;
            hb[2] = (bf16_t)rb[p].z; hb[3] = (bf16_t)rb[p].w;
            *(bf16x4*)&As[buf][row * BKP + lcol] = ha;
            *(bf16x4*)&Bs[buf][row * BKP + lcol] = hb;
        }
    };

    auto compute = [&](int buf) {
        FragU a[4], b[2];
#pragma unroll
        for (int i = 0; i < 4; ++i) {
            // A 16x32 bf16: lanes 0-15 hold K{0-7,16-23}, lanes 16-31 hold K{8-15,24-31}
            const int row = wm * 64 + i * 16 + fm;
            const bf16_t* p = &As[buf][row * BKP + kg * 8];
            a[i].h[0] = *(const bf16x8*)(p);        // K = kg*8 .. +7
            a[i].h[1] = *(const bf16x8*)(p + 16);   // K = 16+kg*8 .. +7
        }
#pragma unroll
        for (int j = 0; j < 2; ++j) {
            // B 32x16 bf16: lanes 0-15 hold K{0-7,16-23}, lanes 16-31 hold K{8-15,24-31}
            const int col = wn * 32 + j * 16 + fm;
            const bf16_t* p = &Bs[buf][col * BKP + kg * 8];
            b[j].h[0] = *(const bf16x8*)(p);
            b[j].h[1] = *(const bf16x8*)(p + 16);
        }
#pragma unroll
        for (int i = 0; i < 4; ++i)
#pragma unroll
            for (int j = 0; j < 2; ++j)
                acc[i][j] = __builtin_amdgcn_wmma_f32_16x16x32_bf16(
                    false, a[i].v, false, b[j].v,
                    (short)0, acc[i][j], false, false);
    };

    gload(0);
    lstore(0);
    __syncthreads();

#pragma unroll 1
    for (int s = 0; s < NSTEPS; ++s) {
        const int buf = s & 1;
        if (s + 1 < NSTEPS) gload(s + 1);     // prefetch next tile while computing
        compute(buf);
        if (s + 1 < NSTEPS) {
            lstore(buf ^ 1);
            asm volatile("s_wait_dscnt 0" ::: "memory");
            __syncthreads();
        }
    }

    // Epilogue: C/D 16x16 f32 layout — VGPR v -> row v + 8*(lane>=16), col = lane&15
    const int crow0 = bm0 + wm * 64 + kg * 8;
    const int ccol0 = bn0 + wn * 32 + fm;
#pragma unroll
    for (int i = 0; i < 4; ++i)
#pragma unroll
        for (int j = 0; j < 2; ++j) {
            float* cp = C + (size_t)(crow0 + i * 16) * N_DIM + (ccol0 + j * 16);
#pragma unroll
            for (int v = 0; v < 8; ++v)
                cp[(size_t)v * N_DIM] = acc[i][j][v];
        }
}

extern "C" void kernel_launch(void* const* d_in, const int* in_sizes, int n_in,
                              void* d_out, int out_size, void* d_ws, size_t ws_size,
                              hipStream_t stream) {
    const float* A = (const float*)d_in[0];   // [R, M, K_local] fp32
    const float* B = (const float*)d_in[1];   // [R, N, K_local] fp32
    float* C = (float*)d_out;                 // [R, M/R, N] == flat [M, N] fp32

    dim3 grid(TILES_M * TILES_N);             // 1024 blocks, swizzled inside kernel
    dim3 block(256);                          // 8 wave32s
    hipLaunchKernelGGL(gemm_rs_bf16_wmma, grid, block, 0, stream, A, B, C);
}
